// ReDAF_78984448573976
// MI455X (gfx1250) — matrix operand order repaired and checked by reading, not compiled, not run on hardware
//
#include <hip/hip_runtime.h>

// ---------------------------------------------------------------------------
// ReDAF forward for gfx1250 (MI455X)
//   out[i,e] = relu( mw[i&1] * relu( (x[i,:]+emb[id[i],:]) . Wt[e,:] + bt[e] ) * zeta )
//   zeta = sigmoid(0.2 * sum(Wrc) + brc)
//
// Fast path (workspace permitting):
//   1) pre-pass: split A=x+emb[id] and Wt into bf16 hi/lo planes in d_ws
//   2) GEMM: double-buffered TDM tensor_load_to_lds (padded bf16 tiles,
//      overlapped with compute via s_wait_tensorcnt 4); bf16x3 WMMA
// Fallback: fused convert-in-loop kernel.
// ---------------------------------------------------------------------------

typedef __attribute__((ext_vector_type(16))) __bf16 v16bf;
typedef __attribute__((ext_vector_type(8)))  float  v8f;
typedef __attribute__((ext_vector_type(4)))  unsigned int v4u;
typedef __attribute__((ext_vector_type(8)))  int v8i;
typedef __attribute__((ext_vector_type(4)))  int v4i;

#define BM 64
#define BN 64
#define BK 32
#define LDSW 40     // fallback LDS row stride (bf16 elems): 80B, conflict-free
#define FBM 64      // fast-path block rows
#define FBN 128     // fast-path block cols (4 subtiles per wave)
#define BK2 64      // fast-path K per barrier (2 slabs of 32)
#define LDSW2 72    // fast-path LDS row stride: 64 data + 8 pad bf16 = 144B

struct U4 { unsigned int a, b, c, d; };       // trivial 16B POD for b128 traffic
union  Frag16 { v16bf v; U4 q[2]; };          // 16 bf16 = 8 VGPRs
union  AccU   { v8f v; float f[8]; };

static __device__ __forceinline__ unsigned short f32_to_bf16_rne(float f) {
    unsigned int u = __builtin_bit_cast(unsigned int, f);
    u += 0x7FFFu + ((u >> 16) & 1u);
    return (unsigned short)(u >> 16);
}
static __device__ __forceinline__ float bf16hi_f32(unsigned short h) {
    unsigned int u = ((unsigned int)h) << 16;
    return __builtin_bit_cast(float, u);
}
static __device__ __forceinline__ void split2(float v0, float v1,
                                              unsigned int& hi, unsigned int& lo) {
    unsigned short h0 = f32_to_bf16_rne(v0);
    unsigned short h1 = f32_to_bf16_rne(v1);
    float r0 = v0 - bf16hi_f32(h0);
    float r1 = v1 - bf16hi_f32(h1);
    unsigned short l0 = f32_to_bf16_rne(r0);
    unsigned short l1 = f32_to_bf16_rne(r1);
    hi = (unsigned int)h0 | ((unsigned int)h1 << 16);
    lo = (unsigned int)l0 | ((unsigned int)l1 << 16);
}

// ---------------- TDM: 2-D tile load, global -> LDS, with LDS padding -------
// Descriptor per CDNA5 ISA §8.3/8.4. pad_interval code 4 = 32 DWORDs (one
// 64-bf16 row), pad_amount code 3 = 4 DWORDs -> LDS row stride 72 bf16.
static __device__ __forceinline__ void tdm_load_2d(const void* gptr, unsigned lds_off,
                                                   unsigned tile_d0, unsigned tile_d1,
                                                   unsigned tensor_d0, unsigned tensor_d1,
                                                   unsigned pad_interval, unsigned pad_amount) {
    unsigned long long ga = (unsigned long long)gptr;
    v4u g0;
    g0[0] = 1u;                                             // count=1, user mode
    g0[1] = lds_off;                                        // lds_addr (bytes)
    g0[2] = (unsigned)ga;                                   // global_addr[31:0]
    g0[3] = (unsigned)((ga >> 32) & 0x01FFFFFFu) | (2u << 30); // ga[56:32] | type=2
    v8i g1;
    g1[0] = (int)((1u << 16)                                // data_size = 2 bytes
                | (1u << 20)                                // pad_enable
                | (pad_interval << 22)
                | (pad_amount << 25));
    g1[1] = (int)((tensor_d0 & 0xFFFFu) << 16);             // tensor_dim0[15:0]
    g1[2] = (int)(((tensor_d0 >> 16) & 0xFFFFu) | ((tensor_d1 & 0xFFFFu) << 16));
    g1[3] = (int)(((tensor_d1 >> 16) & 0xFFFFu) | ((tile_d0 & 0xFFFFu) << 16));
    g1[4] = (int)(tile_d1 & 0xFFFFu);                       // tile_dim1; tile_dim2=0
    g1[5] = (int)tensor_d0;                                 // tensor_dim0_stride[31:0]
    g1[6] = 0;                                              // stride hi / dim1 stride lo
    g1[7] = 0;
    v4i gz = {0, 0, 0, 0};
#if defined(__clang_major__) && (__clang_major__ >= 23)
    v8i gz8 = {0, 0, 0, 0, 0, 0, 0, 0};
    __builtin_amdgcn_tensor_load_to_lds(g0, g1, gz, gz, gz8, 0);
#else
    __builtin_amdgcn_tensor_load_to_lds(g0, g1, gz, gz, 0);
#endif
}

static __device__ __forceinline__ unsigned lds_offset_u32(const void* p) {
    // LDS aperture flat address: low 32 bits are the LDS byte offset
    return (unsigned)(unsigned long long)p;
}

// --------------------------- zeta pre-kernel -------------------------------
__global__ __launch_bounds__(256)
void redaf_zeta_kernel(const float* __restrict__ Wrc,
                       const float* __restrict__ brc,
                       float* __restrict__ zeta_out, int D) {
    __shared__ float red[256];
    float s = 0.f;
    for (int i = threadIdx.x; i < D; i += 256) s += Wrc[i];
    red[threadIdx.x] = s;
    __syncthreads();
    for (int off = 128; off > 0; off >>= 1) {
        if ((int)threadIdx.x < off) red[threadIdx.x] += red[threadIdx.x + off];
        __syncthreads();
    }
    if (threadIdx.x == 0) {
        float z = 0.2f * red[0] + brc[0];
        zeta_out[0] = 1.f / (1.f + expf(-z));
    }
}

// ---------------- pre-pass: A = x + emb[id] -> bf16 hi/lo planes -----------
__global__ void redaf_convertA_kernel(const float* __restrict__ x,
                                      const int* __restrict__ ids,
                                      const float* __restrict__ emb,
                                      unsigned short* __restrict__ Ahi,
                                      unsigned short* __restrict__ Alo, int D) {
    const int row = blockIdx.x;
    const int d0  = threadIdx.x * 8;
    const float* xr = x   + (size_t)row * D + d0;
    const float* er = emb + (size_t)ids[row] * D + d0;
    const float4 a = *(const float4*)xr;
    const float4 b = *(const float4*)(xr + 4);
    const float4 e = *(const float4*)er;
    const float4 f = *(const float4*)(er + 4);
    U4 qh, ql;
    split2(a.x + e.x, a.y + e.y, qh.a, ql.a);
    split2(a.z + e.z, a.w + e.w, qh.b, ql.b);
    split2(b.x + f.x, b.y + f.y, qh.c, ql.c);
    split2(b.z + f.z, b.w + f.w, qh.d, ql.d);
    *(U4*)(Ahi + (size_t)row * D + d0) = qh;
    *(U4*)(Alo + (size_t)row * D + d0) = ql;
}

// ---------------- pre-pass: Wt -> bf16 hi/lo planes ------------------------
__global__ void redaf_convertB_kernel(const float* __restrict__ Wt,
                                      unsigned short* __restrict__ Bhi,
                                      unsigned short* __restrict__ Blo, int D) {
    const int row = blockIdx.x;
    const int d0  = threadIdx.x * 8;
    const float* wr = Wt + (size_t)row * D + d0;
    const float4 a = *(const float4*)wr;
    const float4 b = *(const float4*)(wr + 4);
    U4 qh, ql;
    split2(a.x, a.y, qh.a, ql.a);
    split2(a.z, a.w, qh.b, ql.b);
    split2(b.x, b.y, qh.c, ql.c);
    split2(b.z, b.w, qh.d, ql.d);
    *(U4*)(Bhi + (size_t)row * D + d0) = qh;
    *(U4*)(Blo + (size_t)row * D + d0) = ql;
}

// ---------- fast GEMM: double-buffered TDM staging + bf16x3 WMMA -----------
__global__ __launch_bounds__(256)
void redaf_gemm_tdm_kernel(const unsigned short* __restrict__ Ahi,
                           const unsigned short* __restrict__ Alo,
                           const unsigned short* __restrict__ Bhi,
                           const unsigned short* __restrict__ Blo,
                           const float* __restrict__ modal_w,
                           const float* __restrict__ bt,
                           const float* __restrict__ zeta_p,
                           float* __restrict__ out,
                           int D, int rows) {
    __shared__ alignas(16) unsigned short sAh[2][FBM * LDSW2];
    __shared__ alignas(16) unsigned short sAl[2][FBM * LDSW2];
    __shared__ alignas(16) unsigned short sBh[2][FBN * LDSW2];
    __shared__ alignas(16) unsigned short sBl[2][FBN * LDSW2];

    const int t    = threadIdx.x;
    const int wave = t >> 5;
    const int lane = t & 31;
    const int hl   = lane >> 4;
    const int r    = lane & 15;

    const int blockCol = blockIdx.x;
    const int blockRow = blockIdx.y;

    // wave tiling: 4 row-subtiles x 8 col-subtiles; wave owns 1 row x 4 cols
    const int sm  = wave & 3;
    const int sn0 = (wave >> 2) << 2;

    const size_t aBase = (size_t)(blockRow * FBM) * D;
    const size_t bBase = (size_t)(blockCol * FBN) * D;

    auto issue_tiles = [&](int b, int k0) {
        tdm_load_2d(Ahi + aBase + k0, lds_offset_u32(&sAh[b][0]),
                    BK2, FBM, (unsigned)D, (unsigned)rows, 4u, 3u);
        tdm_load_2d(Alo + aBase + k0, lds_offset_u32(&sAl[b][0]),
                    BK2, FBM, (unsigned)D, (unsigned)rows, 4u, 3u);
        tdm_load_2d(Bhi + bBase + k0, lds_offset_u32(&sBh[b][0]),
                    BK2, FBN, (unsigned)D, (unsigned)D, 4u, 3u);
        tdm_load_2d(Blo + bBase + k0, lds_offset_u32(&sBl[b][0]),
                    BK2, FBN, (unsigned)D, (unsigned)D, 4u, 3u);
    };

    v8f acc[4];
    acc[0] = {};
    acc[1] = {};
    acc[2] = {};
    acc[3] = {};

    const int ksteps = D / BK2;
    if (wave == 0) issue_tiles(0, 0);

    for (int ks = 0; ks < ksteps; ++ks) {
        const int cur = ks & 1;
        if (wave == 0) {
            if (ks + 1 < ksteps) {
                // prefetch next K-step into the other buffer, then wait for
                // the current buffer only (in-order completion, 4 outstanding)
                issue_tiles(cur ^ 1, (ks + 1) * BK2);
                __builtin_amdgcn_s_wait_tensorcnt(4);
            } else {
                __builtin_amdgcn_s_wait_tensorcnt(0);
            }
        }
        __syncthreads();

#pragma unroll
        for (int slab = 0; slab < 2; ++slab) {
            const int kc = slab * 32;
            // A (16x32, MxK): lane row M=r; contiguous K chunk [16*hl,+16)
            Frag16 ah, al;
            {
                const unsigned short* p = &sAh[cur][(sm * 16 + r) * LDSW2 + kc + 16 * hl];
                ah.q[0] = *(const U4*)(p);
                ah.q[1] = *(const U4*)(p + 8);
                const unsigned short* pl = &sAl[cur][(sm * 16 + r) * LDSW2 + kc + 16 * hl];
                al.q[0] = *(const U4*)(pl);
                al.q[1] = *(const U4*)(pl + 8);
            }
#pragma unroll
            for (int j = 0; j < 4; ++j) {
                // B (32x16, KxN): lane col N=r; contiguous K chunk [16*hl,+16)
                Frag16 bh, bl;
                const unsigned short* pb =
                    &sBh[cur][((sn0 + j) * 16 + r) * LDSW2 + kc + 16 * hl];
                bh.q[0] = *(const U4*)(pb);
                bh.q[1] = *(const U4*)(pb + 8);
                const unsigned short* qb =
                    &sBl[cur][((sn0 + j) * 16 + r) * LDSW2 + kc + 16 * hl];
                bl.q[0] = *(const U4*)(qb);
                bl.q[1] = *(const U4*)(qb + 8);
                acc[j] = __builtin_amdgcn_wmma_f32_16x16x32_bf16(false, al.v, false, bh.v,
                                                                 (short)0, acc[j], false, false);
                acc[j] = __builtin_amdgcn_wmma_f32_16x16x32_bf16(false, ah.v, false, bl.v,
                                                                 (short)0, acc[j], false, false);
                acc[j] = __builtin_amdgcn_wmma_f32_16x16x32_bf16(false, ah.v, false, bh.v,
                                                                 (short)0, acc[j], false, false);
            }
        }
        __syncthreads();
    }

    // ---------- epilogue: bias, relu, *mw*zeta, relu ----------
    const float zeta = zeta_p[0];
    const float mw0  = modal_w[0];
    const float mw1  = modal_w[1];
    const int rowbase = blockRow * FBM + sm * 16 + 8 * hl;
#pragma unroll
    for (int j = 0; j < 4; ++j) {
        const int gcol = blockCol * FBN + (sn0 + j) * 16 + r;
        const float bj = bt[gcol];
        AccU u;
        u.v = acc[j];
#pragma unroll
        for (int v = 0; v < 8; ++v) {
            const int grow = rowbase + v;
            const float mw = (grow & 1) ? mw1 : mw0;
            float t0 = u.f[v] + bj;
            t0 = t0 > 0.f ? t0 : 0.f;
            t0 = mw * t0 * zeta;
            t0 = t0 > 0.f ? t0 : 0.f;
            out[(size_t)grow * D + gcol] = t0;
        }
    }
}

// ---------------- fallback GEMM: fused convert-in-loop ---------------------
__global__ __launch_bounds__(256)
void redaf_gemm_kernel(const float* __restrict__ x,
                       const int*   __restrict__ sub_ids,
                       const float* __restrict__ modal_w,
                       const float* __restrict__ emb,
                       const float* __restrict__ Wt,
                       const float* __restrict__ bt,
                       const float* __restrict__ zeta_p,
                       float*       __restrict__ out,
                       int D) {
    __shared__ alignas(16) unsigned short sAh[BM * LDSW];
    __shared__ alignas(16) unsigned short sAl[BM * LDSW];
    __shared__ alignas(16) unsigned short sBh[BN * LDSW];
    __shared__ alignas(16) unsigned short sBl[BN * LDSW];

    const int t    = threadIdx.x;
    const int wave = t >> 5;
    const int lane = t & 31;
    const int hl   = lane >> 4;
    const int r    = lane & 15;

    const int blockCol = blockIdx.x;
    const int blockRow = blockIdx.y;

    const int lrow  = t >> 2;
    const int lquad = t & 3;
    const int growA = blockRow * BM + lrow;
    const int growB = blockCol * BN + lrow;
    const float* xrow = x   + (size_t)growA * D;
    const float* erow = emb + (size_t)sub_ids[growA] * D;
    const float* wrow = Wt  + (size_t)growB * D;

    const int sm  = wave & 3;
    const int sn0 = (wave >> 2) << 1;

    v8f acc0 = {};
    v8f acc1 = {};

    const int ksteps = D / BK;
    for (int ks = 0; ks < ksteps; ++ks) {
        const int d0 = ks * BK + lquad * 8;
        {
            const float4 xa = *(const float4*)(xrow + d0);
            const float4 xb = *(const float4*)(xrow + d0 + 4);
            const float4 ea = *(const float4*)(erow + d0);
            const float4 eb = *(const float4*)(erow + d0 + 4);
            U4 qh, ql;
            split2(xa.x + ea.x, xa.y + ea.y, qh.a, ql.a);
            split2(xa.z + ea.z, xa.w + ea.w, qh.b, ql.b);
            split2(xb.x + eb.x, xb.y + eb.y, qh.c, ql.c);
            split2(xb.z + eb.z, xb.w + eb.w, qh.d, ql.d);
            const int o = lrow * LDSW + lquad * 8;
            *(U4*)(&sAh[o]) = qh;
            *(U4*)(&sAl[o]) = ql;
        }
        {
            const float4 wa = *(const float4*)(wrow + d0);
            const float4 wb = *(const float4*)(wrow + d0 + 4);
            U4 qh, ql;
            split2(wa.x, wa.y, qh.a, ql.a);
            split2(wa.z, wa.w, qh.b, ql.b);
            split2(wb.x, wb.y, qh.c, ql.c);
            split2(wb.z, wb.w, qh.d, ql.d);
            const int o = lrow * LDSW + lquad * 8;
            *(U4*)(&sBh[o]) = qh;
            *(U4*)(&sBl[o]) = ql;
        }
        __syncthreads();

        Frag16 ah, al;
        {
            const unsigned short* p = &sAh[(sm * 16 + r) * LDSW + 16 * hl];
            ah.q[0] = *(const U4*)(p);
            ah.q[1] = *(const U4*)(p + 8);
            const unsigned short* pl = &sAl[(sm * 16 + r) * LDSW + 16 * hl];
            al.q[0] = *(const U4*)(pl);
            al.q[1] = *(const U4*)(pl + 8);
        }
        Frag16 bh0, bl0, bh1, bl1;
        {
            const unsigned short* p0 = &sBh[((sn0)     * 16 + r) * LDSW + 16 * hl];
            bh0.q[0] = *(const U4*)(p0);
            bh0.q[1] = *(const U4*)(p0 + 8);
            const unsigned short* p1 = &sBh[((sn0 + 1) * 16 + r) * LDSW + 16 * hl];
            bh1.q[0] = *(const U4*)(p1);
            bh1.q[1] = *(const U4*)(p1 + 8);
            const unsigned short* q0 = &sBl[((sn0)     * 16 + r) * LDSW + 16 * hl];
            bl0.q[0] = *(const U4*)(q0);
            bl0.q[1] = *(const U4*)(q0 + 8);
            const unsigned short* q1 = &sBl[((sn0 + 1) * 16 + r) * LDSW + 16 * hl];
            bl1.q[0] = *(const U4*)(q1);
            bl1.q[1] = *(const U4*)(q1 + 8);
        }

        acc0 = __builtin_amdgcn_wmma_f32_16x16x32_bf16(false, al.v, false, bh0.v,
                                                       (short)0, acc0, false, false);
        acc0 = __builtin_amdgcn_wmma_f32_16x16x32_bf16(false, ah.v, false, bl0.v,
                                                       (short)0, acc0, false, false);
        acc0 = __builtin_amdgcn_wmma_f32_16x16x32_bf16(false, ah.v, false, bh0.v,
                                                       (short)0, acc0, false, false);
        acc1 = __builtin_amdgcn_wmma_f32_16x16x32_bf16(false, al.v, false, bh1.v,
                                                       (short)0, acc1, false, false);
        acc1 = __builtin_amdgcn_wmma_f32_16x16x32_bf16(false, ah.v, false, bl1.v,
                                                       (short)0, acc1, false, false);
        acc1 = __builtin_amdgcn_wmma_f32_16x16x32_bf16(false, ah.v, false, bh1.v,
                                                       (short)0, acc1, false, false);
        __syncthreads();
    }

    const float zeta = zeta_p[0];
    const float mw0  = modal_w[0];
    const float mw1  = modal_w[1];
    const int gcol0 = blockCol * BN + sn0 * 16 + r;
    const int gcol1 = gcol0 + 16;
    const float b0 = bt[gcol0];
    const float b1 = bt[gcol1];
    AccU u0, u1;
    u0.v = acc0;
    u1.v = acc1;
    const int rowbase = blockRow * BM + sm * 16 + 8 * hl;
#pragma unroll
    for (int v = 0; v < 8; ++v) {
        const int grow = rowbase + v;
        const float mw = (grow & 1) ? mw1 : mw0;
        float t0 = u0.f[v] + b0;
        t0 = t0 > 0.f ? t0 : 0.f;
        t0 = mw * t0 * zeta;
        t0 = t0 > 0.f ? t0 : 0.f;
        out[(size_t)grow * D + gcol0] = t0;
        float t1 = u1.f[v] + b1;
        t1 = t1 > 0.f ? t1 : 0.f;
        t1 = mw * t1 * zeta;
        t1 = t1 > 0.f ? t1 : 0.f;
        out[(size_t)grow * D + gcol1] = t1;
    }
}

// --------------------------------- launch ----------------------------------
extern "C" void kernel_launch(void* const* d_in, const int* in_sizes, int n_in,
                              void* d_out, int out_size, void* d_ws, size_t ws_size,
                              hipStream_t stream) {
    const float* x    = (const float*)d_in[0];
    const int*   ids  = (const int*)  d_in[1];
    const float* mw   = (const float*)d_in[2];
    const float* emb  = (const float*)d_in[3];
    const float* Wt   = (const float*)d_in[4];
    const float* bt   = (const float*)d_in[5];
    const float* Wrc  = (const float*)d_in[6];
    const float* brc  = (const float*)d_in[7];
    float* out  = (float*)d_out;
    float* zeta = (float*)d_ws;

    const int D    = in_sizes[5];   // 2048 (bt length)
    const int rows = in_sizes[1];   // 8192 (B*M)

    redaf_zeta_kernel<<<1, 256, 0, stream>>>(Wrc, brc, zeta, D);

    const size_t planeA = (size_t)rows * (size_t)D;   // elements per A plane
    const size_t planeB = (size_t)D * (size_t)D;      // elements per B plane
    const size_t need   = 256 + 2 * (planeA + planeB) * sizeof(unsigned short);

    if (ws_size >= need && (D % FBN) == 0 && (rows % FBM) == 0 && (D / 8) <= 1024) {
        unsigned short* Ahi = (unsigned short*)((char*)d_ws + 256);
        unsigned short* Alo = Ahi + planeA;
        unsigned short* Bhi = Alo + planeA;
        unsigned short* Blo = Bhi + planeB;
        redaf_convertA_kernel<<<rows, D / 8, 0, stream>>>(x, ids, emb, Ahi, Alo, D);
        redaf_convertB_kernel<<<D, D / 8, 0, stream>>>(Wt, Bhi, Blo, D);
        dim3 fgrid(D / FBN, rows / FBM);
        redaf_gemm_tdm_kernel<<<fgrid, 256, 0, stream>>>(Ahi, Alo, Bhi, Blo,
                                                         mw, bt, zeta, out, D, rows);
    } else {
        dim3 grid(D / BN, rows / BM);
        redaf_gemm_kernel<<<grid, 256, 0, stream>>>(x, ids, mw, emb, Wt, bt, zeta, out, D);
    }
}
